// MaskedSelfAttention_48473000902938
// MI455X (gfx1250) — matrix operand order repaired and checked by reading, not compiled, not run on hardware
//
#include <hip/hip_runtime.h>
#include <cstdint>

// ---------------------------------------------------------------------------
// Masked multi-head self-attention for MI455X (gfx1250), bf16 WMMA pipeline.
// B=4, T=2048, D=1024, H=16, hd=64.  Matrix math on v_wmma_f32_16x16x32_bf16;
// LDS staging via global_load_async_to_lds_b128 (double-buffered in GEMMs).
// ---------------------------------------------------------------------------

#define BATCH 4
#define SEQ   2048
#define DIM   1024
#define NHEAD 16
#define HD    64
#define MROWS (BATCH * SEQ)    // 8192
#define N3    (3 * DIM)        // 3072

typedef __bf16 v16bf __attribute__((ext_vector_type(16)));
typedef float  v8f   __attribute__((ext_vector_type(8)));

union Frag {
    v16bf v;
    uint4 q[2];
};

// ---- async global->LDS machinery (gfx1250), with synchronous fallback ------
#if defined(__has_builtin)
#if __has_builtin(__builtin_amdgcn_global_load_async_to_lds_b128)
#define HAVE_ASYNC_LDS 1
#endif
#endif
#ifndef HAVE_ASYNC_LDS
#define HAVE_ASYNC_LDS 0
#endif

#if HAVE_ASYNC_LDS
typedef int v4i_t __attribute__((vector_size(16)));
typedef v4i_t __attribute__((address_space(1)))* as1_v4i;
typedef v4i_t __attribute__((address_space(3)))* as3_v4i;
// Generic LDS pointers carry the LDS byte offset in their low 32 bits
// (aperture encoding), so truncation yields the addrspace(3) offset.
#define ASYNC_CP16(lds_dst, gsrc)                                              \
    __builtin_amdgcn_global_load_async_to_lds_b128(                            \
        (as1_v4i)(uintptr_t)(const void*)(gsrc),                               \
        (as3_v4i)(unsigned int)(uintptr_t)(void*)(lds_dst), 0, 0)
#if __has_builtin(__builtin_amdgcn_s_wait_asynccnt)
#define ASYNC_WAIT(n) __builtin_amdgcn_s_wait_asynccnt(n)
#else
#define ASYNC_WAIT(n) asm volatile("s_wait_asynccnt %0" ::"n"(n) : "memory")
#endif
#else
#define ASYNC_CP16(lds_dst, gsrc) (*(uint4*)(lds_dst) = *(const uint4*)(gsrc))
#define ASYNC_WAIT(n) do { } while (0)
#endif

__device__ __forceinline__ unsigned short f32_to_bf16(float f) {
    unsigned int u = __float_as_uint(f);
    u += 0x7FFFu + ((u >> 16) & 1u);     // round-to-nearest-even
    return (unsigned short)(u >> 16);
}

__device__ __forceinline__ v8f wmma_bf16(const Frag& a, const Frag& b, v8f c) {
    return __builtin_amdgcn_wmma_f32_16x16x32_bf16(
        /*neg_a=*/false, a.v, /*neg_b=*/false, b.v,
        /*c_mod=*/(short)0, c, /*reuse_a=*/false, /*reuse_b=*/false);
}

// ---------------------------------------------------------------------------
// Prep kernels: f32 -> bf16 convert, and transpose-convert weights to [N,K].
// ---------------------------------------------------------------------------
__global__ __launch_bounds__(256) void cvt_bf16_kernel(
    const float* __restrict__ in, unsigned short* __restrict__ out, int n)
{
    for (int i = blockIdx.x * 256 + threadIdx.x; i < n; i += gridDim.x * 256)
        out[i] = f32_to_bf16(in[i]);
}

// out[nIdx*K + k] = bf16(in[k*N + nIdx]);  in is [K,N] row-major.
__global__ __launch_bounds__(256) void transpose_cvt_kernel(
    const float* __restrict__ in, unsigned short* __restrict__ out, int K, int N)
{
    int total = K * N;
    for (int i = blockIdx.x * 256 + threadIdx.x; i < total; i += gridDim.x * 256) {
        int nIdx = i / K;
        int k    = i - nIdx * K;
        out[i] = f32_to_bf16(in[(size_t)k * N + nIdx]);
    }
}

// ---------------------------------------------------------------------------
// GEMM 1: qkv = xb[8192,1024] @ W_attn + b_attn, routed into Q/K (bf16
// [B,H,T,hd]) and V transposed (bf16 [B,H,hd,T]).
// Block = 256 thr = 8 waves (4x2), block tile 128x128, K tile 64,
// double-buffered async LDS staging.
// ---------------------------------------------------------------------------
#define BK 64
#define LDS_PITCH 72   // 64 bf16 + 8 pad (16B-aligned rows, conflict-reduced)

__global__ __launch_bounds__(256) void qkv_gemm_kernel(
    const unsigned short* __restrict__ A,    // [8192,1024] bf16
    const unsigned short* __restrict__ Bt,   // [3072,1024] bf16 (pre-transposed)
    const float* __restrict__ bias,          // [3072] f32
    unsigned short* __restrict__ Qo,         // [B,H,T,hd] bf16
    unsigned short* __restrict__ Ko,         // [B,H,T,hd] bf16
    unsigned short* __restrict__ Vto)        // [B,H,hd,T] bf16
{
    const int K = DIM;
    int bm = blockIdx.x * 128;
    int bn = blockIdx.y * 128;
    int tid = threadIdx.x;
    int wave = tid >> 5, lane = tid & 31;
    int wm = (wave & 3) * 32, wn = (wave >> 2) * 64;
    int cn = lane & 15, h = lane >> 4;

    __shared__ __align__(16) unsigned short As[2][128 * LDS_PITCH];
    __shared__ __align__(16) unsigned short Bs[2][128 * LDS_PITCH];

    // Stage one 128x64 A tile + 128x64 B tile: 8 async b128 per thread.
    auto stage = [&](int buf, int kb) {
#pragma unroll
        for (int it = 0; it < 4; ++it) {
            int i = tid + it * 256;
            int r = i >> 3, c = i & 7;
            ASYNC_CP16(&As[buf][r * LDS_PITCH + c * 8],
                       A + (size_t)(bm + r) * K + kb + c * 8);
        }
#pragma unroll
        for (int it = 0; it < 4; ++it) {
            int i = tid + it * 256;
            int r = i >> 3, c = i & 7;
            ASYNC_CP16(&Bs[buf][r * LDS_PITCH + c * 8],
                       Bt + (size_t)(bn + r) * K + kb + c * 8);
        }
    };

    v8f acc[2][4];
#pragma unroll
    for (int mi = 0; mi < 2; ++mi)
#pragma unroll
        for (int ni = 0; ni < 4; ++ni)
#pragma unroll
            for (int r = 0; r < 8; ++r) acc[mi][ni][r] = 0.0f;

    const int nk = K / BK;    // 16
    stage(0, 0);
    for (int t = 0; t < nk; ++t) {
        int cur = t & 1;
        if (t + 1 < nk) {
            stage(cur ^ 1, (t + 1) * BK);
            ASYNC_WAIT(8);    // async loads complete in order: tile t is done
        } else {
            ASYNC_WAIT(0);
        }
        __syncthreads();

#pragma unroll
        for (int ks = 0; ks < 2; ++ks) {
            Frag a[2], b[4];
#pragma unroll
            for (int mi = 0; mi < 2; ++mi) {
                const unsigned short* row =
                    &As[cur][(wm + mi * 16 + cn) * LDS_PITCH + ks * 32];
                a[mi].q[0] = *(const uint4*)(row + h * 8);
                a[mi].q[1] = *(const uint4*)(row + 16 + h * 8);
            }
#pragma unroll
            for (int ni = 0; ni < 4; ++ni) {
                const unsigned short* row =
                    &Bs[cur][(wn + ni * 16 + cn) * LDS_PITCH + ks * 32];
                b[ni].q[0] = *(const uint4*)(row + h * 8);
                b[ni].q[1] = *(const uint4*)(row + 16 + h * 8);
            }
#pragma unroll
            for (int mi = 0; mi < 2; ++mi)
#pragma unroll
                for (int ni = 0; ni < 4; ++ni)
                    acc[mi][ni] = wmma_bf16(a[mi], b[ni], acc[mi][ni]);
        }
        __syncthreads();
    }

    // Epilogue: add bias, route to Q / K / V^T as bf16.
#pragma unroll
    for (int mi = 0; mi < 2; ++mi) {
#pragma unroll
        for (int ni = 0; ni < 4; ++ni) {
            int gn = bn + wn + ni * 16 + cn;
            float bv = bias[gn];
            int which = gn >> 10;           // 0=Q, 1=K, 2=V
            int nn = gn & (DIM - 1);
            int hh = nn >> 6;               // head
            int d  = nn & (HD - 1);
#pragma unroll
            for (int r = 0; r < 8; ++r) {
                int gm = bm + wm + mi * 16 + 8 * h + r;
                int bb = gm >> 11;          // batch
                int t  = gm & (SEQ - 1);
                unsigned short val = f32_to_bf16(acc[mi][ni][r] + bv);
                size_t bhI = (size_t)(bb * NHEAD + hh);
                if (which == 0)
                    Qo[(bhI * SEQ + t) * HD + d] = val;
                else if (which == 1)
                    Ko[(bhI * SEQ + t) * HD + d] = val;
                else
                    Vto[(bhI * HD + d) * SEQ + t] = val;
            }
        }
    }
}

// ---------------------------------------------------------------------------
// Flash attention (causal). Block = 128 thr = 4 waves; each wave owns a
// 16-query tile, block covers 64 consecutive queries of one (b,h).
// Keys processed in shared 32-key chunks staged (async) in LDS.
// ---------------------------------------------------------------------------
#define KP 72   // K-tile LDS pitch (64 bf16 + pad)
#define VP 40   // V^T / P LDS pitch (32 bf16 + pad)

__global__ __launch_bounds__(128) void flash_attn_kernel(
    const unsigned short* __restrict__ Qm,   // [B,H,T,hd]
    const unsigned short* __restrict__ Km,   // [B,H,T,hd]
    const unsigned short* __restrict__ Vtm,  // [B,H,hd,T]
    unsigned short* __restrict__ Ob)         // [B*T, D] bf16
{
    int bid = blockIdx.x;
    int qgroup = bid & 31;        // T/64 = 32 groups
    int bh = bid >> 5;            // 0..63
    int base = qgroup * 64;
    int wave = threadIdx.x >> 5;
    int lane = threadIdx.x & 31;
    int cn = lane & 15, h = lane >> 4;
    int qb = base + wave * 16;

    __shared__ __align__(16) unsigned short Ks[32 * KP];      // [key][hd]
    __shared__ __align__(16) unsigned short Vs[64 * VP];      // [d][key]
    __shared__ __align__(16) unsigned short Ps[4][16 * VP];   // per-wave P tile

    const unsigned short* Qbh = Qm  + (size_t)bh * SEQ * HD;
    const unsigned short* Kbh = Km  + (size_t)bh * SEQ * HD;
    const unsigned short* Vbh = Vtm + (size_t)bh * HD * SEQ;

    // Q fragments (A layout): lane holds query row qb+cn, hd split 0..31/32..63.
    Frag qf[2];
    {
        const unsigned short* qrow = Qbh + (size_t)(qb + cn) * HD;
#pragma unroll
        for (int f = 0; f < 2; ++f) {
            qf[f].q[0] = *(const uint4*)(qrow + f * 32 + h * 8);
            qf[f].q[1] = *(const uint4*)(qrow + f * 32 + 16 + h * 8);
        }
    }

    v8f o[4];
#pragma unroll
    for (int i = 0; i < 4; ++i)
#pragma unroll
        for (int r = 0; r < 8; ++r) o[i][r] = 0.0f;
    float sm[8], sl[8];
#pragma unroll
    for (int r = 0; r < 8; ++r) { sm[r] = -1e30f; sl[r] = 0.0f; }

    int kc_end = base + 32;   // last chunk needed by wave 3 (qb=base+48)
    for (int kc = 0; kc <= kc_end; kc += 32) {
        // Async stage: K chunk [32 keys x 64 hd], V^T chunk [64 d x 32 keys].
#pragma unroll
        for (int it = 0; it < 2; ++it) {
            int i = threadIdx.x + it * 128;   // 256 chunks of 16B
            int r = i >> 3, c = i & 7;
            ASYNC_CP16(Ks + r * KP + c * 8,
                       Kbh + (size_t)(kc + r) * HD + c * 8);
        }
#pragma unroll
        for (int it = 0; it < 2; ++it) {
            int i = threadIdx.x + it * 128;   // 256 chunks of 16B
            int r = i >> 2, c = i & 3;
            ASYNC_CP16(Vs + r * VP + c * 8,
                       Vbh + (size_t)r * SEQ + kc + c * 8);
        }
        ASYNC_WAIT(0);
        __syncthreads();

        bool active = (kc <= qb + 15);        // wave-uniform
        if (active) {
            // Scores: two 16(q) x 16(key) tiles, K-dim = hd = 64 (2 wmma each).
            Frag kf[4];
#pragma unroll
            for (int j = 0; j < 2; ++j) {
                const unsigned short* krow = Ks + (j * 16 + cn) * KP;
#pragma unroll
                for (int f = 0; f < 2; ++f) {
                    kf[j * 2 + f].q[0] = *(const uint4*)(krow + f * 32 + h * 8);
                    kf[j * 2 + f].q[1] = *(const uint4*)(krow + f * 32 + 16 + h * 8);
                }
            }
            v8f s[2];
#pragma unroll
            for (int j = 0; j < 2; ++j) {
                v8f c;
#pragma unroll
                for (int r = 0; r < 8; ++r) c[r] = 0.0f;
                c = wmma_bf16(qf[0], kf[j * 2 + 0], c);
                c = wmma_bf16(qf[1], kf[j * 2 + 1], c);
                s[j] = c;
            }

            const float scale = 0.125f;               // 1/sqrt(64)
            bool need_mask = (kc + 31 > qb);
            unsigned short* pw = Ps[wave];
#pragma unroll
            for (int r = 0; r < 8; ++r) {
                float a0 = s[0][r] * scale;
                float a1 = s[1][r] * scale;
                if (need_mask) {
                    int qrow = qb + 8 * h + r;
                    if (kc + cn > qrow)      a0 = -1e30f;
                    if (kc + 16 + cn > qrow) a1 = -1e30f;
                }
                float mx = fmaxf(a0, a1);
#pragma unroll
                for (int d = 1; d < 16; d <<= 1)
                    mx = fmaxf(mx, __shfl_xor(mx, d, 32));
                float mnew  = fmaxf(sm[r], mx);
                float alpha = __expf(sm[r] - mnew);
                sm[r] = mnew;
                float p0 = __expf(a0 - mnew);
                float p1 = __expf(a1 - mnew);
                float ps = p0 + p1;
#pragma unroll
                for (int d = 1; d < 16; d <<= 1)
                    ps += __shfl_xor(ps, d, 32);
                sl[r] = sl[r] * alpha + ps;
#pragma unroll
                for (int i = 0; i < 4; ++i) o[i][r] *= alpha;
                int m = 8 * h + r;
                pw[m * VP + cn]      = f32_to_bf16(p0);
                pw[m * VP + 16 + cn] = f32_to_bf16(p1);
            }
        }
        __syncthreads();   // P tiles visible; V^T still resident

        if (active) {
            // P (A layout) x V (B layout from V^T rows) -> accumulate O.
            Frag pf;
            const unsigned short* prow = Ps[wave] + cn * VP;
            pf.q[0] = *(const uint4*)(prow + h * 8);
            pf.q[1] = *(const uint4*)(prow + 16 + h * 8);
#pragma unroll
            for (int i = 0; i < 4; ++i) {
                Frag vf;
                const unsigned short* vrow = Vs + (i * 16 + cn) * VP;
                vf.q[0] = *(const uint4*)(vrow + h * 8);
                vf.q[1] = *(const uint4*)(vrow + 16 + h * 8);
                o[i] = wmma_bf16(pf, vf, o[i]);
            }
        }
        __syncthreads();   // before next chunk overwrites Ks/Vs
    }

    // Normalize and store bf16 attention output [B*T, D].
    int bb = bh >> 4, head = bh & 15;
#pragma unroll
    for (int i = 0; i < 4; ++i) {
#pragma unroll
        for (int r = 0; r < 8; ++r) {
            int m = 8 * h + r;
            float val = o[i][r] / sl[r];
            size_t idx = (size_t)(bb * SEQ + qb + m) * DIM + head * HD + i * 16 + cn;
            Ob[idx] = f32_to_bf16(val);
        }
    }
}

// ---------------------------------------------------------------------------
// GEMM 2: out = attn_out[8192,1024] @ W_proj + b_proj (f32 output),
// double-buffered async LDS staging.
// ---------------------------------------------------------------------------
__global__ __launch_bounds__(256) void proj_gemm_kernel(
    const unsigned short* __restrict__ A,    // [8192,1024] bf16
    const unsigned short* __restrict__ Bt,   // [1024,1024] bf16 (pre-transposed)
    const float* __restrict__ bias,          // [1024]
    float* __restrict__ out)                 // [8192,1024] f32
{
    const int K = DIM;
    int bm = blockIdx.x * 128;
    int bn = blockIdx.y * 128;
    int tid = threadIdx.x;
    int wave = tid >> 5, lane = tid & 31;
    int wm = (wave & 3) * 32, wn = (wave >> 2) * 64;
    int cn = lane & 15, h = lane >> 4;

    __shared__ __align__(16) unsigned short As[2][128 * LDS_PITCH];
    __shared__ __align__(16) unsigned short Bs[2][128 * LDS_PITCH];

    auto stage = [&](int buf, int kb) {
#pragma unroll
        for (int it = 0; it < 4; ++it) {
            int i = tid + it * 256;
            int r = i >> 3, c = i & 7;
            ASYNC_CP16(&As[buf][r * LDS_PITCH + c * 8],
                       A + (size_t)(bm + r) * K + kb + c * 8);
        }
#pragma unroll
        for (int it = 0; it < 4; ++it) {
            int i = tid + it * 256;
            int r = i >> 3, c = i & 7;
            ASYNC_CP16(&Bs[buf][r * LDS_PITCH + c * 8],
                       Bt + (size_t)(bn + r) * K + kb + c * 8);
        }
    };

    v8f acc[2][4];
#pragma unroll
    for (int mi = 0; mi < 2; ++mi)
#pragma unroll
        for (int ni = 0; ni < 4; ++ni)
#pragma unroll
            for (int r = 0; r < 8; ++r) acc[mi][ni][r] = 0.0f;

    const int nk = K / BK;    // 16
    stage(0, 0);
    for (int t = 0; t < nk; ++t) {
        int cur = t & 1;
        if (t + 1 < nk) {
            stage(cur ^ 1, (t + 1) * BK);
            ASYNC_WAIT(8);
        } else {
            ASYNC_WAIT(0);
        }
        __syncthreads();

#pragma unroll
        for (int ks = 0; ks < 2; ++ks) {
            Frag a[2], b[4];
#pragma unroll
            for (int mi = 0; mi < 2; ++mi) {
                const unsigned short* row =
                    &As[cur][(wm + mi * 16 + cn) * LDS_PITCH + ks * 32];
                a[mi].q[0] = *(const uint4*)(row + h * 8);
                a[mi].q[1] = *(const uint4*)(row + 16 + h * 8);
            }
#pragma unroll
            for (int ni = 0; ni < 4; ++ni) {
                const unsigned short* row =
                    &Bs[cur][(wn + ni * 16 + cn) * LDS_PITCH + ks * 32];
                b[ni].q[0] = *(const uint4*)(row + h * 8);
                b[ni].q[1] = *(const uint4*)(row + 16 + h * 8);
            }
#pragma unroll
            for (int mi = 0; mi < 2; ++mi)
#pragma unroll
                for (int ni = 0; ni < 4; ++ni)
                    acc[mi][ni] = wmma_bf16(a[mi], b[ni], acc[mi][ni]);
        }
        __syncthreads();
    }

#pragma unroll
    for (int mi = 0; mi < 2; ++mi) {
#pragma unroll
        for (int ni = 0; ni < 4; ++ni) {
            int gn = bn + wn + ni * 16 + cn;
            float bv = bias[gn];
#pragma unroll
            for (int r = 0; r < 8; ++r) {
                int gm = bm + wm + mi * 16 + 8 * h + r;
                out[(size_t)gm * DIM + gn] = acc[mi][ni][r] + bv;
            }
        }
    }
}

// ---------------------------------------------------------------------------
// Host-side launcher.  Workspace layout (bf16 buffers), ~92.3 MB total:
//   xb[8192,1024] | WaT[3072,1024] | WpT[1024,1024] | Q | K | V^T | attn_out
// ---------------------------------------------------------------------------
extern "C" void kernel_launch(void* const* d_in, const int* in_sizes, int n_in,
                              void* d_out, int out_size, void* d_ws, size_t ws_size,
                              hipStream_t stream) {
    const float* x      = (const float*)d_in[0];   // [4,2048,1024]
    const float* W_attn = (const float*)d_in[1];   // [1024,3072]
    const float* b_attn = (const float*)d_in[2];   // [3072]
    const float* W_proj = (const float*)d_in[3];   // [1024,1024]
    const float* b_proj = (const float*)d_in[4];   // [1024]
    float* out = (float*)d_out;

    char* ws = (char*)d_ws;
    size_t off = 0;
    unsigned short* xb  = (unsigned short*)(ws + off); off += (size_t)MROWS * DIM * 2;
    unsigned short* WaT = (unsigned short*)(ws + off); off += (size_t)N3 * DIM * 2;
    unsigned short* WpT = (unsigned short*)(ws + off); off += (size_t)DIM * DIM * 2;
    unsigned short* Qb  = (unsigned short*)(ws + off); off += (size_t)MROWS * DIM * 2;
    unsigned short* Kb  = (unsigned short*)(ws + off); off += (size_t)MROWS * DIM * 2;
    unsigned short* Vt  = (unsigned short*)(ws + off); off += (size_t)MROWS * DIM * 2;
    unsigned short* AOb = (unsigned short*)(ws + off); off += (size_t)MROWS * DIM * 2;

    cvt_bf16_kernel<<<4096, 256, 0, stream>>>(x, xb, MROWS * DIM);
    transpose_cvt_kernel<<<2048, 256, 0, stream>>>(W_attn, WaT, DIM, N3);
    transpose_cvt_kernel<<<1024, 256, 0, stream>>>(W_proj, WpT, DIM, DIM);

    dim3 g1(MROWS / 128, N3 / 128);   // 64 x 24
    qkv_gemm_kernel<<<g1, 256, 0, stream>>>(xb, WaT, b_attn, Qb, Kb, Vt);

    flash_attn_kernel<<<BATCH * NHEAD * (SEQ / 64), 128, 0, stream>>>(Qb, Kb, Vt, AOb);

    dim3 g2(MROWS / 128, DIM / 128);  // 64 x 8
    proj_gemm_kernel<<<g2, 256, 0, stream>>>(AOb, WpT, b_proj, out);
}
